// AxialLinearTransformerLayer_37555194036783
// MI455X (gfx1250) — compile-verified, matrix operand order checked by reading, not run on hardware
//
#include <hip/hip_runtime.h>

// Problem constants (from reference).
#define EE 64
#define HH 4
#define BB 2
#define RR 1225
#define CC 512

// ---------------------------------------------------------------------------
// bf16 / WMMA helpers (gfx1250 wave32, v_wmma_f32_16x16x32_bf16)
// ---------------------------------------------------------------------------
typedef __bf16 bf16_t;
typedef bf16_t v16bf __attribute__((ext_vector_type(16)));
typedef bf16_t v8bf  __attribute__((ext_vector_type(8)));
typedef float  v8f   __attribute__((ext_vector_type(8)));

static __device__ __forceinline__ bf16_t f2bf(float f) { return (bf16_t)f; }
static __device__ __forceinline__ float bf2f(bf16_t b) { return (float)b; }
static __device__ __forceinline__ float elu1(float v) {  // elu(x)+1
  return v > 0.f ? v + 1.f : __expf(v);
}

// A-frag (16x32, row=M) and B-frag (32x16, row=N) share the same per-lane
// layout: lane L in [0,16): row=L, k in {0..7,16..23}; lane L+16: k in
// {8..15,24..31}. Source must be row-major with k contiguous, stride mult of 8.
static __device__ __forceinline__ v16bf load_frag(const bf16_t* base, int row, int stride) {
  const int lane = threadIdx.x & 31;
  const int m = lane & 15, hf = lane >> 4;
  const bf16_t* p = base + (row + m) * stride + hf * 8;
  v8bf lo = *(const v8bf*)(p);
  v8bf hi = *(const v8bf*)(p + 16);
  v16bf r;
  #pragma unroll
  for (int i = 0; i < 8; ++i) { r[i] = lo[i]; r[i + 8] = hi[i]; }
  return r;
}
static __device__ __forceinline__ v8f wmma_bf16(v16bf a, v16bf b, v8f c) {
  return __builtin_amdgcn_wmma_f32_16x16x32_bf16(false, a, false, b, (short)0, c, false, false);
}
#define V8F_Z {0.f,0.f,0.f,0.f,0.f,0.f,0.f,0.f}
// D-frag element i of lane L: m = i + 8*(L>>4), n = L&15.

// ---------------------------------------------------------------------------
// LDS layouts (bytes). All bf16 fragment bases / row strides are 16B aligned.
// ---------------------------------------------------------------------------
// Kernel 1: row attention (one (b,r) per block; C=512 in 4 tiles of 128)
constexpr int K1_XT   = 0;                        // 128*65 f32 (x tile / out accum)
constexpr int K1_XLN  = K1_XT   + 128*65*4;       // 128*72 bf16 (LN(x) / o)
constexpr int K1_WV   = K1_XLN  + 128*72*2;       // 64*72 bf16 (wv, n-major)
constexpr int K1_WO   = K1_WV   + 64*72*2;        // 64*72 bf16
constexpr int K1_WQ   = K1_WO   + 64*72*2;        // 4*64 f32
constexpr int K1_WK   = K1_WQ   + 4*64*4;
constexpr int K1_BQ   = K1_WK   + 4*64*4;         // 4 f32
constexpr int K1_BK   = K1_BQ   + 16;
constexpr int K1_BV   = K1_BK   + 16;             // 64 f32
constexpr int K1_BO   = K1_BV   + 256;
constexpr int K1_LG   = K1_BO   + 256;
constexpr int K1_LB   = K1_LG   + 256;
constexpr int K1_QB   = K1_LB   + 256;            // 512*4 f32 (q' per pos/head)
constexpr int K1_KTVA = K1_QB   + 512*4*4;        // 16*136 bf16 (k'^T, M-padded)
constexpr int K1_VST  = K1_KTVA + 16*136*2;       // 64*136 bf16 (V^T, n-major)
constexpr int K1_SQ   = K1_VST  + 64*136*2;       // 4 f32
constexpr int K1_SK   = K1_SQ   + 16;             // 4 f32
constexpr int K1_KTVB = K1_SK   + 16;             // 4*64 f32
constexpr int K1_SZ   = K1_KTVB + 4*64*4;         // 104256 B

// Kernel 2a: column stats (one (b,c) per block; R streamed in 10 tiles)
constexpr int K2A_XT   = 0;
constexpr int K2A_XLN  = K2A_XT   + 128*65*4;
constexpr int K2A_WV   = K2A_XLN  + 128*72*2;
constexpr int K2A_WQ   = K2A_WV   + 64*72*2;
constexpr int K2A_WK   = K2A_WQ   + 4*64*4;
constexpr int K2A_BQ   = K2A_WK   + 4*64*4;
constexpr int K2A_BK   = K2A_BQ   + 16;
constexpr int K2A_BV   = K2A_BK   + 16;
constexpr int K2A_LG   = K2A_BV   + 256;
constexpr int K2A_LB   = K2A_LG   + 256;
constexpr int K2A_KTVA = K2A_LB   + 256;
constexpr int K2A_VST  = K2A_KTVA + 16*136*2;
constexpr int K2A_SQ   = K2A_VST  + 64*136*2;
constexpr int K2A_SK   = K2A_SQ   + 16;
constexpr int K2A_KTVB = K2A_SK   + 16;
constexpr int K2A_SZ   = K2A_KTVB + 4*64*4;       // 86592 B

// Kernel 2b: column apply (128-position c-tile per block)
constexpr int K2B_XT  = 0;
constexpr int K2B_XLN = K2B_XT  + 128*65*4;
constexpr int K2B_WO  = K2B_XLN + 128*72*2;
constexpr int K2B_WQ  = K2B_WO  + 64*72*2;
constexpr int K2B_BQ  = K2B_WQ  + 1024;
constexpr int K2B_BO  = K2B_BQ  + 16;
constexpr int K2B_LG  = K2B_BO  + 256;
constexpr int K2B_LB  = K2B_LG  + 256;
constexpr int K2B_SZ  = K2B_LB  + 256;            // 62736 B

// Kernel 3: FFN (128-position c-tile per block)
constexpr int K3_XT  = 0;                         // 128*65 f32 (h tile / accum)
constexpr int K3_XLN = K3_XT  + 128*65*4;         // 128*72 bf16
constexpr int K3_W1  = K3_XLN + 128*72*2;         // 256*72 bf16 (w1, n-major)
constexpr int K3_W2  = K3_W1  + 256*72*2;         // 64*264 bf16 (w2, n-major)
constexpr int K3_G1  = K3_W2  + 64*264*2;         // 128*264 bf16 (gelu acts)
constexpr int K3_B1  = K3_G1  + 128*264*2;        // 256 f32
constexpr int K3_B2  = K3_B1  + 1024;             // 64 f32
constexpr int K3_LG  = K3_B2  + 256;
constexpr int K3_LB  = K3_LG  + 256;
constexpr int K3_SZ  = K3_LB  + 256;              // 191744 B

// ---------------------------------------------------------------------------
// Kernel 1: row attention  h = x + RowAttn(LN(x))   (x is (B,E,R,C))
// ---------------------------------------------------------------------------
__global__ __launch_bounds__(256) void k_row_attn(
    const float* __restrict__ x,
    const float* __restrict__ wq, const float* __restrict__ bq,
    const float* __restrict__ wk, const float* __restrict__ bk,
    const float* __restrict__ wv, const float* __restrict__ bv,
    const float* __restrict__ wo, const float* __restrict__ bo,
    const float* __restrict__ lng, const float* __restrict__ lnb,
    float* __restrict__ hbuf)
{
  extern __shared__ char smem[];
  float*  xt   = (float*)(smem + K1_XT);
  bf16_t* xln  = (bf16_t*)(smem + K1_XLN);
  bf16_t* wvB  = (bf16_t*)(smem + K1_WV);
  bf16_t* woB  = (bf16_t*)(smem + K1_WO);
  float*  wqS  = (float*)(smem + K1_WQ);
  float*  wkS  = (float*)(smem + K1_WK);
  float*  bqS  = (float*)(smem + K1_BQ);
  float*  bkS  = (float*)(smem + K1_BK);
  float*  bvS  = (float*)(smem + K1_BV);
  float*  boS  = (float*)(smem + K1_BO);
  float*  lgS  = (float*)(smem + K1_LG);
  float*  lbS  = (float*)(smem + K1_LB);
  float*  qbuf = (float*)(smem + K1_QB);
  bf16_t* ktvA = (bf16_t*)(smem + K1_KTVA);
  bf16_t* Vst  = (bf16_t*)(smem + K1_VST);
  float*  sq   = (float*)(smem + K1_SQ);
  float*  sk   = (float*)(smem + K1_SK);
  float*  ktvb = (float*)(smem + K1_KTVB);

  const int tid  = threadIdx.x;
  const int wave = tid >> 5;
  const int lane = tid & 31;
  const int nn   = lane & 15;
  const int hf   = lane >> 4;
  const int b = blockIdx.x / RR;
  const int r = blockIdx.x % RR;
  const size_t xbr = (size_t)b * EE * RR * CC + (size_t)r * CC;

  for (int i = tid; i < 64 * 64; i += 256) {
    int n = i >> 6, k = i & 63;
    wvB[n * 72 + k] = f2bf(wv[i]);
    woB[n * 72 + k] = f2bf(wo[i]);
  }
  if (tid < 256) { wqS[tid] = wq[tid]; wkS[tid] = wk[tid]; }
  if (tid < 4)  { bqS[tid] = bq[tid]; bkS[tid] = bk[tid]; sq[tid] = 0.f; sk[tid] = 0.f; }
  if (tid < 64) { bvS[tid] = bv[tid]; boS[tid] = bo[tid]; lgS[tid] = lng[tid]; lbS[tid] = lnb[tid]; }
  for (int i = tid; i < 16 * 136; i += 256) ktvA[i] = f2bf(0.f);

  v8f acc_ktv = V8F_Z;

  // ---- Phase B: LN fused with q/k projection, v GEMM, ktv GEMM ----
  for (int t = 0; t < CC / 128; ++t) {
    const int c0 = t * 128;
    __syncthreads();
    for (int i = tid; i < 128 * 64; i += 256) {
      int e = i >> 7, cl = i & 127;
      xt[cl * 65 + e] = x[xbr + (size_t)e * RR * CC + c0 + cl];
    }
    __syncthreads();
    if (tid < 128) {
      int cl = tid;
      float m = 0.f;
      for (int e = 0; e < 64; ++e) m += xt[cl * 65 + e];
      m *= (1.f / 64.f);
      float v = 0.f;
      for (int e = 0; e < 64; ++e) { float d = xt[cl * 65 + e] - m; v += d * d; }
      float inv = rsqrtf(v * (1.f / 64.f) + 1e-5f);
      float qh[4], kh[4];
      #pragma unroll
      for (int h = 0; h < 4; ++h) { qh[h] = bqS[h]; kh[h] = bkS[h]; }
      for (int e = 0; e < 64; ++e) {
        float z = (xt[cl * 65 + e] - m) * inv * lgS[e] + lbS[e];
        xln[cl * 72 + e] = f2bf(z);
        #pragma unroll
        for (int h = 0; h < 4; ++h) { qh[h] += wqS[h * 64 + e] * z; kh[h] += wkS[h * 64 + e] * z; }
      }
      #pragma unroll
      for (int h = 0; h < 4; ++h) {
        float q = elu1(qh[h]), k = elu1(kh[h]);
        qbuf[(c0 + cl) * 4 + h] = q;
        atomicAdd(&sq[h], q);
        atomicAdd(&sk[h], k);
        ktvA[h * 136 + cl] = f2bf(k);
      }
    }
    __syncthreads();
    // V = LN(x) @ wv^T + bv, stored n-major (as B operand for the ktv GEMM)
    for (int wt = wave; wt < 32; wt += 8) {
      int mt = wt >> 2, nt = wt & 3;
      v8f acc = V8F_Z;
      #pragma unroll
      for (int ks = 0; ks < 2; ++ks) {
        v16bf a  = load_frag(xln + ks * 32, mt * 16, 72);
        v16bf bm = load_frag(wvB + ks * 32, nt * 16, 72);
        acc = wmma_bf16(a, bm, acc);
      }
      int n = nt * 16 + nn;
      float bias = bvS[n];
      #pragma unroll
      for (int i = 0; i < 8; ++i) {
        int cl = mt * 16 + i + 8 * hf;
        Vst[n * 136 + cl] = f2bf(acc[i] + bias);
      }
    }
    __syncthreads();
    if (wave < 4) {  // ktv (M=16 pad, N=64, K=128): waves 0..3 own one N-tile
      #pragma unroll
      for (int ks = 0; ks < 4; ++ks) {
        v16bf a  = load_frag(ktvA + ks * 32, 0, 136);
        v16bf bm = load_frag(Vst + ks * 32, wave * 16, 136);
        acc_ktv = wmma_bf16(a, bm, acc_ktv);
      }
    }
  }
  __syncthreads();
  if (wave < 4 && hf == 0) {
    #pragma unroll
    for (int i = 0; i < 4; ++i) ktvb[i * 64 + wave * 16 + nn] = acc_ktv[i];
  }
  __syncthreads();

  // ---- Phase C: o = (q'/mean_q) * (ktv/sum_k), out = o @ wo^T + bo + x ----
  for (int t = 0; t < CC / 128; ++t) {
    const int c0 = t * 128;
    __syncthreads();
    for (int i = tid; i < 128 * 64; i += 256) {
      int e = i >> 7, cl = i & 127;
      xt[cl * 65 + e] = x[xbr + (size_t)e * RR * CC + c0 + cl];
    }
    if (tid < 128) {
      int cl = tid;
      #pragma unroll
      for (int h = 0; h < 4; ++h) {
        float qn = qbuf[(c0 + cl) * 4 + h] * ((float)CC / sq[h]);
        float ik = 1.f / sk[h];
        #pragma unroll
        for (int d = 0; d < 16; ++d)
          xln[cl * 72 + h * 16 + d] = f2bf(qn * ktvb[h * 64 + h * 16 + d] * ik);
      }
    }
    __syncthreads();
    for (int wt = wave; wt < 32; wt += 8) {
      int mt = wt >> 2, nt = wt & 3;
      v8f acc = V8F_Z;
      #pragma unroll
      for (int ks = 0; ks < 2; ++ks) {
        v16bf a  = load_frag(xln + ks * 32, mt * 16, 72);
        v16bf bm = load_frag(woB + ks * 32, nt * 16, 72);
        acc = wmma_bf16(a, bm, acc);
      }
      int e = nt * 16 + nn;
      float bias = boS[e];
      #pragma unroll
      for (int i = 0; i < 8; ++i) {
        int cl = mt * 16 + i + 8 * hf;
        xt[cl * 65 + e] += acc[i] + bias;
      }
    }
    __syncthreads();
    const size_t hb = (((size_t)b * RR + r) * CC + c0) * EE;
    for (int i = tid; i < 128 * 64; i += 256)
      hbuf[hb + i] = xt[(i >> 6) * 65 + (i & 63)];
  }
}

// ---------------------------------------------------------------------------
// Kernel 2a: column-attention stats per (b,c,h): {sum q', sum k', ktv[16]}
// ---------------------------------------------------------------------------
__global__ __launch_bounds__(256) void k_col_stats(
    const float* __restrict__ hbuf,
    const float* __restrict__ wq, const float* __restrict__ bq,
    const float* __restrict__ wk, const float* __restrict__ bk,
    const float* __restrict__ wv, const float* __restrict__ bv,
    const float* __restrict__ lng, const float* __restrict__ lnb,
    float* __restrict__ cstats)
{
  extern __shared__ char smem[];
  float*  xt   = (float*)(smem + K2A_XT);
  bf16_t* xln  = (bf16_t*)(smem + K2A_XLN);
  bf16_t* wvB  = (bf16_t*)(smem + K2A_WV);
  float*  wqS  = (float*)(smem + K2A_WQ);
  float*  wkS  = (float*)(smem + K2A_WK);
  float*  bqS  = (float*)(smem + K2A_BQ);
  float*  bkS  = (float*)(smem + K2A_BK);
  float*  bvS  = (float*)(smem + K2A_BV);
  float*  lgS  = (float*)(smem + K2A_LG);
  float*  lbS  = (float*)(smem + K2A_LB);
  bf16_t* ktvA = (bf16_t*)(smem + K2A_KTVA);
  bf16_t* Vst  = (bf16_t*)(smem + K2A_VST);
  float*  sq   = (float*)(smem + K2A_SQ);
  float*  sk   = (float*)(smem + K2A_SK);
  float*  ktvb = (float*)(smem + K2A_KTVB);

  const int tid  = threadIdx.x;
  const int wave = tid >> 5;
  const int lane = tid & 31;
  const int nn   = lane & 15;
  const int hf   = lane >> 4;
  const int b = blockIdx.x / CC;
  const int c = blockIdx.x % CC;

  for (int i = tid; i < 64 * 64; i += 256) {
    int n = i >> 6, k = i & 63;
    wvB[n * 72 + k] = f2bf(wv[i]);
  }
  if (tid < 256) { wqS[tid] = wq[tid]; wkS[tid] = wk[tid]; }
  if (tid < 4)  { bqS[tid] = bq[tid]; bkS[tid] = bk[tid]; sq[tid] = 0.f; sk[tid] = 0.f; }
  if (tid < 64) { bvS[tid] = bv[tid]; lgS[tid] = lng[tid]; lbS[tid] = lnb[tid]; }
  for (int i = tid; i < 16 * 136; i += 256) ktvA[i] = f2bf(0.f);

  v8f acc_ktv = V8F_Z;
  const int RT = (RR + 127) / 128;
  for (int t = 0; t < RT; ++t) {
    const int r0 = t * 128;
    __syncthreads();
    for (int i = tid; i < 128 * 64; i += 256) {
      int rl = i >> 6, e = i & 63, r = r0 + rl;
      xt[rl * 65 + e] = (r < RR) ? hbuf[(((size_t)b * RR + r) * CC + c) * EE + e] : 0.f;
    }
    __syncthreads();
    if (tid < 128) {
      int rl = tid;
      if (r0 + rl < RR) {
        float m = 0.f;
        for (int e = 0; e < 64; ++e) m += xt[rl * 65 + e];
        m *= (1.f / 64.f);
        float v = 0.f;
        for (int e = 0; e < 64; ++e) { float d = xt[rl * 65 + e] - m; v += d * d; }
        float inv = rsqrtf(v * (1.f / 64.f) + 1e-5f);
        float qh[4], kh[4];
        #pragma unroll
        for (int h = 0; h < 4; ++h) { qh[h] = bqS[h]; kh[h] = bkS[h]; }
        for (int e = 0; e < 64; ++e) {
          float z = (xt[rl * 65 + e] - m) * inv * lgS[e] + lbS[e];
          xln[rl * 72 + e] = f2bf(z);
          #pragma unroll
          for (int h = 0; h < 4; ++h) { qh[h] += wqS[h * 64 + e] * z; kh[h] += wkS[h * 64 + e] * z; }
        }
        #pragma unroll
        for (int h = 0; h < 4; ++h) {
          float q = elu1(qh[h]), k = elu1(kh[h]);
          atomicAdd(&sq[h], q);
          atomicAdd(&sk[h], k);
          ktvA[h * 136 + rl] = f2bf(k);
        }
      } else {
        for (int e = 0; e < 64; ++e) xln[rl * 72 + e] = f2bf(0.f);
        #pragma unroll
        for (int h = 0; h < 4; ++h) ktvA[h * 136 + rl] = f2bf(0.f);
      }
    }
    __syncthreads();
    for (int wt = wave; wt < 32; wt += 8) {
      int mt = wt >> 2, nt = wt & 3;
      v8f acc = V8F_Z;
      #pragma unroll
      for (int ks = 0; ks < 2; ++ks) {
        v16bf a  = load_frag(xln + ks * 32, mt * 16, 72);
        v16bf bm = load_frag(wvB + ks * 32, nt * 16, 72);
        acc = wmma_bf16(a, bm, acc);
      }
      int n = nt * 16 + nn;
      float bias = bvS[n];
      #pragma unroll
      for (int i = 0; i < 8; ++i) {
        int rl = mt * 16 + i + 8 * hf;
        Vst[n * 136 + rl] = f2bf(acc[i] + bias);
      }
    }
    __syncthreads();
    if (wave < 4) {
      #pragma unroll
      for (int ks = 0; ks < 4; ++ks) {
        v16bf a  = load_frag(ktvA + ks * 32, 0, 136);
        v16bf bm = load_frag(Vst + ks * 32, wave * 16, 136);
        acc_ktv = wmma_bf16(a, bm, acc_ktv);
      }
    }
  }
  __syncthreads();
  if (wave < 4 && hf == 0) {
    #pragma unroll
    for (int i = 0; i < 4; ++i) ktvb[i * 64 + wave * 16 + nn] = acc_ktv[i];
  }
  __syncthreads();
  if (tid < HH * 18) {
    int h = tid / 18, j = tid - h * 18;
    float v = (j == 0) ? sq[h] : (j == 1) ? sk[h] : ktvb[h * 64 + h * 16 + (j - 2)];
    cstats[(size_t)(b * CC + c) * (HH * 18) + tid] = v;
  }
}

// ---------------------------------------------------------------------------
// Kernel 2b: column-attention apply: h += o @ wo^T + bo (in place)
// ---------------------------------------------------------------------------
__global__ __launch_bounds__(256) void k_col_apply(
    float* hbuf,
    const float* __restrict__ wq, const float* __restrict__ bq,
    const float* __restrict__ wo, const float* __restrict__ bo,
    const float* __restrict__ lng, const float* __restrict__ lnb,
    const float* __restrict__ cstats)
{
  extern __shared__ char smem[];
  float*  xt  = (float*)(smem + K2B_XT);
  bf16_t* xln = (bf16_t*)(smem + K2B_XLN);
  bf16_t* woB = (bf16_t*)(smem + K2B_WO);
  float*  wqS = (float*)(smem + K2B_WQ);
  float*  bqS = (float*)(smem + K2B_BQ);
  float*  boS = (float*)(smem + K2B_BO);
  float*  lgS = (float*)(smem + K2B_LG);
  float*  lbS = (float*)(smem + K2B_LB);

  const int tid  = threadIdx.x;
  const int wave = tid >> 5;
  const int lane = tid & 31;
  const int nn   = lane & 15;
  const int hf   = lane >> 4;
  const int b   = blockIdx.x / (RR * 4);
  const int rem = blockIdx.x % (RR * 4);
  const int r   = rem >> 2;
  const int c0  = (rem & 3) * 128;
  const size_t base = (((size_t)b * RR + r) * CC + c0) * EE;

  for (int i = tid; i < 64 * 64; i += 256) {
    int n = i >> 6, k = i & 63;
    woB[n * 72 + k] = f2bf(wo[i]);
  }
  if (tid < 256) wqS[tid] = wq[tid];
  if (tid < 4)  bqS[tid] = bq[tid];
  if (tid < 64) { boS[tid] = bo[tid]; lgS[tid] = lng[tid]; lbS[tid] = lnb[tid]; }

  for (int i = tid; i < 128 * 64; i += 256)
    xt[(i >> 6) * 65 + (i & 63)] = hbuf[base + i];
  __syncthreads();
  if (tid < 128) {
    int cl = tid;
    float m = 0.f;
    for (int e = 0; e < 64; ++e) m += xt[cl * 65 + e];
    m *= (1.f / 64.f);
    float v = 0.f;
    for (int e = 0; e < 64; ++e) { float d = xt[cl * 65 + e] - m; v += d * d; }
    float inv = rsqrtf(v * (1.f / 64.f) + 1e-5f);
    float qh[4];
    #pragma unroll
    for (int h = 0; h < 4; ++h) qh[h] = bqS[h];
    for (int e = 0; e < 64; ++e) {
      float z = (xt[cl * 65 + e] - m) * inv * lgS[e] + lbS[e];
      #pragma unroll
      for (int h = 0; h < 4; ++h) qh[h] += wqS[h * 64 + e] * z;
    }
    const float* st = cstats + (size_t)(b * CC + c0 + cl) * (HH * 18);
    #pragma unroll
    for (int h = 0; h < 4; ++h) {
      float qn = elu1(qh[h]) * ((float)RR / st[h * 18 + 0]);
      float ik = 1.f / st[h * 18 + 1];
      #pragma unroll
      for (int d = 0; d < 16; ++d)
        xln[cl * 72 + h * 16 + d] = f2bf(qn * st[h * 18 + 2 + d] * ik);
    }
  }
  __syncthreads();
  for (int wt = wave; wt < 32; wt += 8) {
    int mt = wt >> 2, nt = wt & 3;
    v8f acc = V8F_Z;
    #pragma unroll
    for (int ks = 0; ks < 2; ++ks) {
      v16bf a  = load_frag(xln + ks * 32, mt * 16, 72);
      v16bf bm = load_frag(woB + ks * 32, nt * 16, 72);
      acc = wmma_bf16(a, bm, acc);
    }
    int e = nt * 16 + nn;
    float bias = boS[e];
    #pragma unroll
    for (int i = 0; i < 8; ++i) {
      int cl = mt * 16 + i + 8 * hf;
      xt[cl * 65 + e] += acc[i] + bias;
    }
  }
  __syncthreads();
  for (int i = tid; i < 128 * 64; i += 256)
    hbuf[base + i] = xt[(i >> 6) * 65 + (i & 63)];
}

// ---------------------------------------------------------------------------
// Kernel 3: FFN + residual + transpose to (B,E,R,C)
// ---------------------------------------------------------------------------
__global__ __launch_bounds__(256) void k_ffn(
    const float* __restrict__ hbuf,
    const float* __restrict__ w1, const float* __restrict__ b1,
    const float* __restrict__ w2, const float* __restrict__ b2,
    const float* __restrict__ lng, const float* __restrict__ lnb,
    float* __restrict__ out)
{
  extern __shared__ char smem[];
  float*  xt  = (float*)(smem + K3_XT);
  bf16_t* xln = (bf16_t*)(smem + K3_XLN);
  bf16_t* w1B = (bf16_t*)(smem + K3_W1);
  bf16_t* w2B = (bf16_t*)(smem + K3_W2);
  bf16_t* g1  = (bf16_t*)(smem + K3_G1);
  float*  b1S = (float*)(smem + K3_B1);
  float*  b2S = (float*)(smem + K3_B2);
  float*  lgS = (float*)(smem + K3_LG);
  float*  lbS = (float*)(smem + K3_LB);

  const int tid  = threadIdx.x;
  const int wave = tid >> 5;
  const int lane = tid & 31;
  const int nn   = lane & 15;
  const int hf   = lane >> 4;
  const int b   = blockIdx.x / (RR * 4);
  const int rem = blockIdx.x % (RR * 4);
  const int r   = rem >> 2;
  const int c0  = (rem & 3) * 128;
  const size_t base = (((size_t)b * RR + r) * CC + c0) * EE;

  for (int i = tid; i < 256 * 64; i += 256) {
    int n = i >> 6, k = i & 63;
    w1B[n * 72 + k] = f2bf(w1[i]);
  }
  for (int i = tid; i < 64 * 256; i += 256) {
    int n = i >> 8, k = i & 255;
    w2B[n * 264 + k] = f2bf(w2[i]);
  }
  if (tid < 256) b1S[tid] = b1[tid];
  if (tid < 64) { b2S[tid] = b2[tid]; lgS[tid] = lng[tid]; lbS[tid] = lnb[tid]; }
  for (int i = tid; i < 128 * 64; i += 256)
    xt[(i >> 6) * 65 + (i & 63)] = hbuf[base + i];
  __syncthreads();
  if (tid < 128) {
    int cl = tid;
    float m = 0.f;
    for (int e = 0; e < 64; ++e) m += xt[cl * 65 + e];
    m *= (1.f / 64.f);
    float v = 0.f;
    for (int e = 0; e < 64; ++e) { float d = xt[cl * 65 + e] - m; v += d * d; }
    float inv = rsqrtf(v * (1.f / 64.f) + 1e-5f);
    for (int e = 0; e < 64; ++e)
      xln[cl * 72 + e] = f2bf((xt[cl * 65 + e] - m) * inv * lgS[e] + lbS[e]);
  }
  __syncthreads();
  // GEMM1 (128x256x64) + exact GELU
  for (int wt = wave; wt < 128; wt += 8) {
    int mt = wt >> 4, nt = wt & 15;
    v8f acc = V8F_Z;
    #pragma unroll
    for (int ks = 0; ks < 2; ++ks) {
      v16bf a  = load_frag(xln + ks * 32, mt * 16, 72);
      v16bf bm = load_frag(w1B + ks * 32, nt * 16, 72);
      acc = wmma_bf16(a, bm, acc);
    }
    int n = nt * 16 + nn;
    float bias = b1S[n];
    #pragma unroll
    for (int i = 0; i < 8; ++i) {
      int cl = mt * 16 + i + 8 * hf;
      float z = acc[i] + bias;
      g1[cl * 264 + n] = f2bf(0.5f * z * (1.f + erff(z * 0.70710678118654752f)));
    }
  }
  __syncthreads();
  // GEMM2 (128x64x256) + b2 + residual
  for (int wt = wave; wt < 32; wt += 8) {
    int mt = wt >> 2, nt = wt & 3;
    v8f acc = V8F_Z;
    #pragma unroll
    for (int ks = 0; ks < 8; ++ks) {
      v16bf a  = load_frag(g1 + ks * 32, mt * 16, 264);
      v16bf bm = load_frag(w2B + ks * 32, nt * 16, 264);
      acc = wmma_bf16(a, bm, acc);
    }
    int e = nt * 16 + nn;
    float bias = b2S[e];
    #pragma unroll
    for (int i = 0; i < 8; ++i) {
      int cl = mt * 16 + i + 8 * hf;
      xt[cl * 65 + e] += acc[i] + bias;
    }
  }
  __syncthreads();
  // transposed, coalesced store to (B,E,R,C)
  for (int i = tid; i < 64 * 128; i += 256) {
    int e = i >> 7, cl = i & 127;
    out[(((size_t)b * EE + e) * RR + r) * CC + c0 + cl] = xt[cl * 65 + e];
  }
}

// ---------------------------------------------------------------------------
extern "C" void kernel_launch(void* const* d_in, const int* in_sizes, int n_in,
                              void* d_out, int out_size, void* d_ws, size_t ws_size,
                              hipStream_t stream) {
  (void)in_sizes; (void)n_in; (void)out_size; (void)ws_size;
  const float* x      = (const float*)d_in[0];
  const float* row_wq = (const float*)d_in[1];
  const float* row_bq = (const float*)d_in[2];
  const float* row_wk = (const float*)d_in[3];
  const float* row_bk = (const float*)d_in[4];
  const float* row_wv = (const float*)d_in[5];
  const float* row_bv = (const float*)d_in[6];
  const float* row_wo = (const float*)d_in[7];
  const float* row_bo = (const float*)d_in[8];
  const float* col_wq = (const float*)d_in[9];
  const float* col_bq = (const float*)d_in[10];
  const float* col_wk = (const float*)d_in[11];
  const float* col_bk = (const float*)d_in[12];
  const float* col_wv = (const float*)d_in[13];
  const float* col_bv = (const float*)d_in[14];
  const float* col_wo = (const float*)d_in[15];
  const float* col_bo = (const float*)d_in[16];
  const float* rn_g   = (const float*)d_in[17];
  const float* rn_b   = (const float*)d_in[18];
  const float* cn_g   = (const float*)d_in[19];
  const float* cn_b   = (const float*)d_in[20];
  const float* fn_g   = (const float*)d_in[21];
  const float* fn_b   = (const float*)d_in[22];
  const float* w1     = (const float*)d_in[23];
  const float* b1     = (const float*)d_in[24];
  const float* w2     = (const float*)d_in[25];
  const float* b2     = (const float*)d_in[26];

  float* hbuf   = (float*)d_ws;
  float* cstats = hbuf + (size_t)BB * RR * CC * EE;

  dim3 blk(256);
  k_row_attn<<<dim3(BB * RR), blk, K1_SZ, stream>>>(
      x, row_wq, row_bq, row_wk, row_bk, row_wv, row_bv, row_wo, row_bo,
      rn_g, rn_b, hbuf);
  k_col_stats<<<dim3(BB * CC), blk, K2A_SZ, stream>>>(
      hbuf, col_wq, col_bq, col_wk, col_bk, col_wv, col_bv, cn_g, cn_b, cstats);
  k_col_apply<<<dim3(BB * RR * (CC / 128)), blk, K2B_SZ, stream>>>(
      hbuf, col_wq, col_bq, col_wo, col_bo, cn_g, cn_b, cstats);
  k_ffn<<<dim3(BB * RR * (CC / 128)), blk, K3_SZ, stream>>>(
      hbuf, w1, b1, w2, b2, fn_g, fn_b, (float*)d_out);
}
